// SmoothConv2d_57483842290268
// MI455X (gfx1250) — matrix operand order repaired and checked by reading, not compiled, not run on hardware
//
#include <hip/hip_runtime.h>

typedef __attribute__((ext_vector_type(16))) _Float16 v16h;
typedef __attribute__((ext_vector_type(8)))  _Float16 v8h;
typedef __attribute__((ext_vector_type(8)))  float    v8f;

#define N_IMG   8
#define C_IN    256
#define H_IN    52
#define W_IN    52
#define FN_OUT  512
#define OH      52
#define OW      52
#define HP      54
#define WP      54
#define SPATIAL (OH * OW)                 // 2704
#define M_TOTAL (N_IMG * SPATIAL)         // 21632
#define K_TOTAL (C_IN * 9)                // 2304

#define BM 128
#define BN 128
#define BK 32
#define BKP 40                            // padded LDS row (80B) -> conflict-free frag loads
#define KSTEPS (K_TOTAL / BK)             // 72

#define PAD_ELEMS (N_IMG * HP * WP * C_IN)   // 5,971,968 f16
#define WT_ELEMS  (K_TOTAL * FN_OUT)         // 1,179,648 f16

// ---------------------------------------------------------------------------
// Prep 1: int8 quantize (scale 1/0.05=20, round-to-even, clamp) then /2, into
// zero-padded NHWC f16: pad[n][y][x][c], y,x in [0,54)
// ---------------------------------------------------------------------------
__global__ void quant_pad_kernel(const float* __restrict__ in,
                                 _Float16* __restrict__ pad) {
    int idx = blockIdx.x * 256 + threadIdx.x;
    if (idx >= PAD_ELEMS) return;
    int c = idx & (C_IN - 1);
    int r = idx >> 8;
    int x = r % WP; r /= WP;
    int y = r % HP;
    int n = r / HP;
    float v = 0.f;
    if (y >= 1 && y <= H_IN && x >= 1 && x <= W_IN) {
        float t = in[((n * C_IN + c) * H_IN + (y - 1)) * W_IN + (x - 1)] * 20.f;
        t = rintf(t);
        t = fminf(127.f, fmaxf(-128.f, t));
        v = t * 0.5f;   // / INPUT_SMOOTH
    }
    pad[idx] = (_Float16)v;
}

// ---------------------------------------------------------------------------
// Prep 2: weight [C*9, FN] fp32 (K ordered c-major) -> wtT [FN][K] f16 with
// K reordered as (tap, c) so B tiles load exactly like A tiles.
// ---------------------------------------------------------------------------
__global__ void weight_prep_kernel(const float* __restrict__ w,
                                   _Float16* __restrict__ wt) {
    int idx = blockIdx.x * 256 + threadIdx.x;
    if (idx >= WT_ELEMS) return;
    int k  = idx % K_TOTAL;
    int fn = idx / K_TOTAL;
    int t  = k >> 8;          // tap 0..8
    int c  = k & (C_IN - 1);
    wt[idx] = (_Float16)w[(c * 9 + t) * FN_OUT + fn];
}

// ---------------------------------------------------------------------------
// Main: implicit-im2col GEMM, 8 waves, BM x BN = 128x128, double-buffered LDS,
// 8 x v_wmma_f32_16x16x32_f16 per wave per k-step.
// ---------------------------------------------------------------------------
__launch_bounds__(256)
__global__ void conv_gemm_kernel(const _Float16* __restrict__ act,
                                 const _Float16* __restrict__ wt,
                                 const float* __restrict__ bias,
                                 float* __restrict__ out) {
    __shared__ __align__(16) _Float16 sA[2][BM][BKP];
    __shared__ __align__(16) _Float16 sB[2][BN][BKP];

    const int tid   = threadIdx.x;
    const int lane  = tid & 31;
    const int wv    = tid >> 5;
    const int waveM = wv >> 2;      // 0..1  (64 rows each)
    const int waveN = wv & 3;       // 0..3  (32 cols each)
    const int hl    = lane >> 4;    // lane half
    const int lcol  = lane & 15;

    const int mBase = blockIdx.x * BM;
    const int nBase = blockIdx.y * BN;

    // global->LDS staging: 128 rows x 4 chunks of 8 halves; 2 slots per thread
    const int rowT  = tid >> 2;           // 0..63
    const int chunk = (tid & 3) * 8;      // 0,8,16,24

    size_t rowAddrA[2];
    const _Float16* rowB[2];
#pragma unroll
    for (int s = 0; s < 2; ++s) {
        int m  = mBase + rowT + s * 64;
        int ni = m / SPATIAL;
        int sp = m - ni * SPATIAL;
        int oy = sp / OW;
        int ox = sp - oy * OW;
        rowAddrA[s] = (size_t)((ni * HP + oy) * WP + ox) * C_IN;  // tap (0,0) base
        rowB[s]     = wt + (size_t)(nBase + rowT + s * 64) * K_TOTAL;
    }

    v8f acc[4][2] = {};

    // ---- prologue: k-step 0 (tap 0, cbase 0) ----
    uint4 ra[2], rb[2];
#pragma unroll
    for (int s = 0; s < 2; ++s) {
        ra[s] = *(const uint4*)(act + rowAddrA[s] + chunk);
        rb[s] = *(const uint4*)(rowB[s] + chunk);
    }
#pragma unroll
    for (int s = 0; s < 2; ++s) {
        *(uint4*)&sA[0][rowT + s * 64][chunk] = ra[s];
        *(uint4*)&sB[0][rowT + s * 64][chunk] = rb[s];
    }
    __syncthreads();

    for (int kk = 0; kk < KSTEPS; ++kk) {
        const int  cur  = kk & 1;
        const bool more = (kk + 1) < KSTEPS;

        if (more) {                      // prefetch next k-step into registers
            int kn  = kk + 1;
            int tap = kn >> 3;
            int cb  = (kn & 7) * 32;
            int fy  = tap / 3, fx = tap - fy * 3;
            size_t tapOff = (size_t)((fy * WP + fx) * C_IN + cb + chunk);
#pragma unroll
            for (int s = 0; s < 2; ++s) {
                ra[s] = *(const uint4*)(act + rowAddrA[s] + tapOff);
                rb[s] = *(const uint4*)(rowB[s] + (size_t)kn * BK + chunk);
            }
        }

        // ---- fragment loads per documented 16-bit A 16x32 VGPR layout ----
        v16h afrag[4], bfrag[2];
#pragma unroll
        for (int fm = 0; fm < 4; ++fm) {
            int mr = waveM * 64 + fm * 16 + lcol;
            union { v16h v; v8h p[2]; } u;
            u.p[0] = *(const v8h*)&sA[cur][mr][hl * 8];        // K 0..7 / 8..15
            u.p[1] = *(const v8h*)&sA[cur][mr][16 + hl * 8];   // K 16..23 / 24..31
            afrag[fm] = u.v;
        }
#pragma unroll
        for (int fn = 0; fn < 2; ++fn) {
            int nr = waveN * 32 + fn * 16 + lcol;
            union { v16h v; v8h p[2]; } ub;
            ub.p[0] = *(const v8h*)&sB[cur][nr][hl * 8];        // K 0..7 / 8..15
            ub.p[1] = *(const v8h*)&sB[cur][nr][16 + hl * 8];   // K 16..23 / 24..31
            bfrag[fn] = ub.v;
        }

#pragma unroll
        for (int fm = 0; fm < 4; ++fm)
#pragma unroll
            for (int fn = 0; fn < 2; ++fn)
                acc[fm][fn] = __builtin_amdgcn_wmma_f32_16x16x32_f16(
                    false, afrag[fm], false, bfrag[fn],
                    (short)0, acc[fm][fn], false, false);

        if (more) {
            const int nxt = cur ^ 1;
#pragma unroll
            for (int s = 0; s < 2; ++s) {
                *(uint4*)&sA[nxt][rowT + s * 64][chunk] = ra[s];
                *(uint4*)&sB[nxt][rowT + s * 64][chunk] = rb[s];
            }
            __syncthreads();
        }
    }

    // ---- epilogue: + 8*bias, *0.5, round-to-even, clamp; stage as i8 ----
    __syncthreads();
    signed char* stage = (signed char*)&sA[0][0][0];   // 128x128 i8 = 16 KB
#pragma unroll
    for (int fm = 0; fm < 4; ++fm) {
#pragma unroll
        for (int fn = 0; fn < 2; ++fn) {
            int ncol = waveN * 32 + fn * 16 + lcol;
            float b8 = 8.f * bias[nBase + ncol];       // BIAS_SLICE*INPUT_SLICE
#pragma unroll
            for (int v = 0; v < 8; ++v) {
                float t = (acc[fm][fn][v] + b8) * 0.5f;  // p = 16/(4*8)
                t = rintf(t);
                t = fminf(127.f, fmaxf(-128.f, t));
                int mloc = waveM * 64 + fm * 16 + hl * 8 + v;
                stage[ncol * BM + mloc] = (signed char)t;
            }
        }
    }
    __syncthreads();

    // ---- coalesced NCHW writeout: per wave, 16 fn-rows, float4 per lane ----
#pragma unroll
    for (int i = 0; i < 16; ++i) {
        int ncol = wv * 16 + i;
        int mloc = lane * 4;
        const signed char* p = &stage[ncol * BM + mloc];
        float4 f = make_float4((float)p[0], (float)p[1], (float)p[2], (float)p[3]);
        int m  = mBase + mloc;
        int ni = m / SPATIAL;
        int sp = m - ni * SPATIAL;
        size_t off = (size_t)ni * (FN_OUT * SPATIAL)
                   + (size_t)(nBase + ncol) * SPATIAL + sp;
        *(float4*)(out + off) = f;
    }
}

// ---------------------------------------------------------------------------
extern "C" void kernel_launch(void* const* d_in, const int* in_sizes, int n_in,
                              void* d_out, int out_size, void* d_ws, size_t ws_size,
                              hipStream_t stream) {
    const float* input  = (const float*)d_in[0];
    const float* weight = (const float*)d_in[1];
    const float* bias   = (const float*)d_in[2];
    float* out = (float*)d_out;

    _Float16* padAct = (_Float16*)d_ws;                                   // 11.94 MB
    _Float16* wtT    = (_Float16*)((char*)d_ws + (size_t)PAD_ELEMS * 2);  // +2.36 MB

    quant_pad_kernel<<<(PAD_ELEMS + 255) / 256, 256, 0, stream>>>(input, padAct);
    weight_prep_kernel<<<(WT_ELEMS + 255) / 256, 256, 0, stream>>>(weight, wtT);

    dim3 grid(M_TOTAL / BM, FN_OUT / BN);   // 169 x 4, exact
    conv_gemm_kernel<<<grid, 256, 0, stream>>>(padAct, wtT, bias, out);
}
